// StochasticResidualQuantizer_58480274702750
// MI455X (gfx1250) — compile-verified, matrix operand order checked by reading, not run on hardware
//
#include <hip/hip_runtime.h>

// ---------------- problem constants ----------------
#define NROW   32768          // B*H*W = 32*32*32
#define CDIM   64             // embedding dim
#define KCB    1024           // num embeddings
#define NUMQ   4
#define BATCH  32
#define HW     1024           // 32*32

// output layout (floats) in d_out, concatenated in return order
#define OUT_QUANT_OFF 0
#define OUT_LOSS_OFF  2097152
#define OUT_IDX_OFF   2097153
#define OUT_SOFT_OFF  (2097153 + NUMQ * NROW)     // 2,228,225
#define SOFT_STRIDE   ((size_t)NROW * KCB)        // 33,554,432

// workspace layout (bytes, 16B aligned)
#define WS_RESID_F32   0u               // N*C f32   = 8 MB
#define WS_QUANT_F32   8388608u         // N*C f32   = 8 MB
#define WS_RESID_BF16  16777216u        // N*C bf16  = 4 MB
#define WS_CB_BF16     20971520u        // K*C bf16  = 128 KB
#define WS_R2          21102592u        // N f32     = 128 KB
#define WS_CB2         21233664u        // K f32     = 4 KB
#define WS_LOSSROW     21237760u        // N f32     = 128 KB

typedef __attribute__((ext_vector_type(16))) __bf16 v16bf;
typedef __attribute__((ext_vector_type(8)))  __bf16 v8bf;
typedef __attribute__((ext_vector_type(8)))  float  v8f;

// ---------------- init kernels ----------------

// x [B,C,H,W] -> residual [N,C] (f32 + bf16), zero quantized accumulator
__global__ void init_resid(const float* __restrict__ x,
                           float* __restrict__ resid,
                           __bf16* __restrict__ residb,
                           float* __restrict__ quant) {
  int idx = blockIdx.x * blockDim.x + threadIdx.x;
  if (idx >= NROW * CDIM) return;
  int n = idx >> 6;            // / 64
  int c = idx & 63;
  int b = n >> 10;             // / (H*W)
  int rem = n & 1023;          // h*W + w
  float v = x[(((size_t)b * CDIM + c) << 10) + rem];
  resid[idx]  = v;
  residb[idx] = (__bf16)v;
  quant[idx]  = 0.0f;
}

// codebook -> bf16, per-code squared norm
__global__ void init_cb(const float* __restrict__ cbf,
                        __bf16* __restrict__ cbb,
                        float* __restrict__ cb2) {
  int k = blockIdx.x * blockDim.x + threadIdx.x;
  if (k >= KCB) return;
  float s = 0.0f;
  #pragma unroll 8
  for (int c = 0; c < CDIM; ++c) {
    float v = cbf[k * CDIM + c];
    cbb[k * CDIM + c] = (__bf16)v;
    s += v * v;
  }
  cb2[k] = s;
}

// per-row squared norm of residual; zero per-row loss accumulator
__global__ void init_r2(const float* __restrict__ resid,
                        float* __restrict__ r2,
                        float* __restrict__ lossrow) {
  int n = blockIdx.x * blockDim.x + threadIdx.x;
  if (n >= NROW) return;
  float s = 0.0f;
  #pragma unroll 8
  for (int c = 0; c < CDIM; ++c) {
    float v = resid[n * CDIM + c];
    s += v * v;
  }
  r2[n] = s;
  lossrow[n] = 0.0f;
}

// ---------------- stage GEMM: logits = -(||r||^2 + ||c||^2 - 2 r.c) ----------------
// grid = NROW/16 blocks of 256 threads (8 waves). Block handles row-tile m0 = 16 rows
// over all 1024 codes; each wave owns 8 contiguous 16-col tiles (wave*128).
__global__ void gemm_logits(const __bf16* __restrict__ rb,
                            const __bf16* __restrict__ cb,
                            const float* __restrict__ r2,
                            const float* __restrict__ cb2,
                            float* __restrict__ logits) {
  const int lane = threadIdx.x & 31;
  const int wave = threadIdx.x >> 5;
  const int m0   = blockIdx.x << 4;
  const int hsel = (lane < 16) ? 0 : 8;   // K-half select per 16-bit A/B VGPR layout
  const int mrow = m0 + (lane & 15);

  // A: 16x32 bf16 tiles for k-steps 0 and 32 (ISA layout: lanes0-15 K{0..7,16..23},
  // lanes16-31 K{8..15,24..31}); each chunk is a 16B aligned v8bf load.
  v16bf a0, a1;
  {
    const __bf16* pr = rb + (size_t)mrow * CDIM;
    v8bf c0 = *(const v8bf*)(pr + 0  + hsel);
    v8bf c1 = *(const v8bf*)(pr + 16 + hsel);
    v8bf c2 = *(const v8bf*)(pr + 32 + hsel);
    v8bf c3 = *(const v8bf*)(pr + 48 + hsel);
    #pragma unroll
    for (int i = 0; i < 8; ++i) {
      a0[i] = c0[i]; a0[i + 8] = c1[i];
      a1[i] = c2[i]; a1[i + 8] = c3[i];
    }
  }

  float r2m[8];
  #pragma unroll
  for (int v = 0; v < 8; ++v) r2m[v] = r2[m0 + v + hsel];

  #pragma unroll 2
  for (int t = 0; t < 8; ++t) {
    const int n0   = (wave << 7) + (t << 4);
    const int ncol = n0 + (lane & 15);
    const __bf16* pc = cb + (size_t)ncol * CDIM;
    v16bf b0, b1;
    {
      v8bf d0 = *(const v8bf*)(pc + 0  + hsel);
      v8bf d1 = *(const v8bf*)(pc + 16 + hsel);
      v8bf d2 = *(const v8bf*)(pc + 32 + hsel);
      v8bf d3 = *(const v8bf*)(pc + 48 + hsel);
      #pragma unroll
      for (int i = 0; i < 8; ++i) {
        b0[i] = d0[i]; b0[i + 8] = d1[i];
        b1[i] = d2[i]; b1[i + 8] = d3[i];
      }
    }

    v8f acc = {};
    acc = __builtin_amdgcn_wmma_f32_16x16x32_bf16(false, a0, false, b0,
                                                  (short)0, acc, false, false);
    acc = __builtin_amdgcn_wmma_f32_16x16x32_bf16(false, a1, false, b1,
                                                  (short)0, acc, false, false);

    const float c2 = cb2[ncol];
    // C/D layout: VGPR v, lane l -> M = v + (l<16?0:8), N = l&15
    #pragma unroll
    for (int v = 0; v < 8; ++v) {
      float logit = -(r2m[v] + c2 - 2.0f * acc[v]);
      logits[(size_t)(m0 + v + hsel) * KCB + ncol] = logit;
    }
  }
}

// ---------------- softmax + gumbel sample + residual update ----------------
// one block (256 threads) per row; logits in-place -> probs
__global__ void softmax_sample_update(float* __restrict__ probs,
                                      const float* __restrict__ cbf,
                                      float* __restrict__ resid,
                                      __bf16* __restrict__ residb,
                                      float* __restrict__ quant,
                                      float* __restrict__ r2,
                                      float* __restrict__ lossrow,
                                      float* __restrict__ idx_out,
                                      int stage) {
  const int row = blockIdx.x;
  const int tid = threadIdx.x;
  float* rowp = probs + (size_t)row * KCB;

  __shared__ float smax[256];
  __shared__ float sred[256];
  __shared__ int   sidx[256];
  __shared__ int   schosen;
  __shared__ float ssq[64];

  float l[4];
  float lmax = -1e30f, best = -1e30f;
  int bidx = 0;
  #pragma unroll
  for (int j = 0; j < 4; ++j) {
    int k = tid + j * 256;
    float lv = rowp[k];
    l[j] = lv;
    lmax = fmaxf(lmax, lv);
    // hash-based gumbel noise (stand-in for jax categorical)
    unsigned h = (unsigned)(row * KCB + k) ^ (0x9E3779B9u * (unsigned)(stage + 1));
    h ^= h >> 16; h *= 0x7feb352du; h ^= h >> 15; h *= 0x846ca68bu; h ^= h >> 16;
    float u = ((h >> 8) + 1u) * (1.0f / 16777217.0f);
    float g = -logf(-logf(u));
    float sc = lv + g;
    if (sc > best) { best = sc; bidx = k; }
  }
  smax[tid] = lmax; sred[tid] = best; sidx[tid] = bidx;
  __syncthreads();
  for (int s = 128; s > 0; s >>= 1) {
    if (tid < s) {
      smax[tid] = fmaxf(smax[tid], smax[tid + s]);
      if (sred[tid + s] > sred[tid]) { sred[tid] = sred[tid + s]; sidx[tid] = sidx[tid + s]; }
    }
    __syncthreads();
  }
  const float rowmax = smax[0];
  if (tid == 0) { schosen = sidx[0]; idx_out[row] = (float)sidx[0]; }
  __syncthreads();

  float lsum = 0.0f;
  #pragma unroll
  for (int j = 0; j < 4; ++j) { l[j] = expf(l[j] - rowmax); lsum += l[j]; }
  sred[tid] = lsum;
  __syncthreads();
  for (int s = 128; s > 0; s >>= 1) {
    if (tid < s) sred[tid] += sred[tid + s];
    __syncthreads();
  }
  const float inv = 1.0f / sred[0];
  #pragma unroll
  for (int j = 0; j < 4; ++j) rowp[tid + j * 256] = l[j] * inv;

  // residual / quantized / loss update (forward value of q + 0.25e loss is
  // 1.25 * mean((cb[idx]-r)^2) = 1.25 * mean(new_residual^2))
  const int chosen = schosen;
  if (tid < CDIM) {
    float cv = cbf[(size_t)chosen * CDIM + tid];
    size_t o = (size_t)row * CDIM + tid;
    float rn = resid[o] - cv;
    resid[o]  = rn;
    residb[o] = (__bf16)rn;
    quant[o] += cv;
    ssq[tid] = rn * rn;
  }
  __syncthreads();
  if (tid < 32) ssq[tid] += ssq[tid + 32];
  __syncthreads();
  if (tid == 0) {
    float s = 0.0f;
    #pragma unroll
    for (int i = 0; i < 32; ++i) s += ssq[i];
    r2[row] = s;                 // next stage's ||r||^2
    lossrow[row] += s;           // deterministic loss partial
  }
}

// ---------------- deterministic loss reduction (single block) ----------------
__global__ void reduce_loss(const float* __restrict__ lossrow, float* __restrict__ loss_slot) {
  __shared__ float s[256];
  float a = 0.0f;
  for (int i = threadIdx.x; i < NROW; i += 256) a += lossrow[i];
  s[threadIdx.x] = a;
  __syncthreads();
  for (int k = 128; k > 0; k >>= 1) {
    if (threadIdx.x < k) s[threadIdx.x] += s[threadIdx.x + k];
    __syncthreads();
  }
  if (threadIdx.x == 0) loss_slot[0] = s[0] * (1.25f / 2097152.0f);
}

// ---------------- quantized [N,C] -> [B,C,H,W] writeback ----------------
__global__ void finalize(const float* __restrict__ quant, float* __restrict__ out) {
  int idx = blockIdx.x * blockDim.x + threadIdx.x;
  if (idx >= NROW * CDIM) return;
  int n = idx >> 6;
  int c = idx & 63;
  int b = n >> 10;
  int rem = n & 1023;
  out[(((size_t)b * CDIM + c) << 10) + rem] = quant[idx];
}

// ---------------- launch ----------------
extern "C" void kernel_launch(void* const* d_in, const int* in_sizes, int n_in,
                              void* d_out, int out_size, void* d_ws, size_t ws_size,
                              hipStream_t stream) {
  (void)in_sizes; (void)n_in; (void)out_size; (void)ws_size;
  const float* x   = (const float*)d_in[0];   // [32,64,32,32]
  const float* cbf = (const float*)d_in[1];   // [1024,64]
  float* out = (float*)d_out;
  char*  ws  = (char*)d_ws;

  float*  resid   = (float*)(ws + WS_RESID_F32);
  float*  quant   = (float*)(ws + WS_QUANT_F32);
  __bf16* residb  = (__bf16*)(ws + WS_RESID_BF16);
  __bf16* cbb     = (__bf16*)(ws + WS_CB_BF16);
  float*  r2      = (float*)(ws + WS_R2);
  float*  cb2     = (float*)(ws + WS_CB2);
  float*  lossrow = (float*)(ws + WS_LOSSROW);

  init_resid<<<(NROW * CDIM + 255) / 256, 256, 0, stream>>>(x, resid, residb, quant);
  init_cb<<<(KCB + 255) / 256, 256, 0, stream>>>(cbf, cbb, cb2);
  init_r2<<<(NROW + 255) / 256, 256, 0, stream>>>(resid, r2, lossrow);

  for (int s = 0; s < NUMQ; ++s) {
    float* soft = out + OUT_SOFT_OFF + (size_t)s * SOFT_STRIDE;
    float* idxs = out + OUT_IDX_OFF + (size_t)s * NROW;
    gemm_logits<<<NROW / 16, 256, 0, stream>>>(residb, cbb, r2, cb2, soft);
    softmax_sample_update<<<NROW, 256, 0, stream>>>(soft, cbf, resid, residb,
                                                    quant, r2, lossrow, idxs, s);
  }

  reduce_loss<<<1, 256, 0, stream>>>(lossrow, out + OUT_LOSS_OFF);
  finalize<<<(NROW * CDIM + 255) / 256, 256, 0, stream>>>(quant, out);
}
